// EncoderRNN_54262616818269
// MI455X (gfx1250) — compile-verified, matrix operand order checked by reading, not run on hardware
//
#include <hip/hip_runtime.h>
#include <hip/hip_bf16.h>
#include <math.h>

// Problem dims (fixed by reference)
#define B_  64
#define T_  512
#define I_  512
#define H_  1024
#define G3  (3 * H_)     // 3072 gate columns
#define M_  (B_ * T_)    // 32768 rows for input projection

typedef __attribute__((ext_vector_type(16))) __bf16 v16bf;
typedef __attribute__((ext_vector_type(8)))  float  v8f;

union Frag {
    uint4 q[2];   // two 16-byte chunks = 16 bf16
    v16bf v;
};

__device__ __forceinline__ unsigned short f32_to_bf16_rne(float f) {
    unsigned u = __float_as_uint(f);
    u += 0x7FFFu + ((u >> 16) & 1u);   // round-to-nearest-even
    return (unsigned short)(u >> 16);
}

// ---------------------------------------------------------------------------
// fp32 -> bf16 conversion (grid-stride)
// ---------------------------------------------------------------------------
__global__ void cvt_f32_bf16_kernel(const float* __restrict__ in,
                                    unsigned short* __restrict__ out, int n) {
    int i = blockIdx.x * blockDim.x + threadIdx.x;
    int stride = gridDim.x * blockDim.x;
    for (; i < n; i += stride) out[i] = f32_to_bf16_rne(in[i]);
}

// ---------------------------------------------------------------------------
// Big parallel GEMM: out[M x 3072] = A[M x K](bf16) * W[3072 x K]^T(bf16) + bias
// WG = 256 threads (8 waves, wave32). WG tile 64(M) x 128(N).
// wave w: m-subtile = (w>>1), n-half = (w&1) covering 4 n-subtiles of 16.
// ---------------------------------------------------------------------------
__global__ __launch_bounds__(256) void gemm_bias_kernel(
    const unsigned short* __restrict__ A,    // [M x K] bf16 row-major
    const unsigned short* __restrict__ W,    // [3072 x K] bf16 row-major (=B^T)
    const float* __restrict__ bias,          // [3072]
    float* __restrict__ out,                 // [M x 3072] fp32
    int K)
{
    const int lane = threadIdx.x & 31;
    const int wave = threadIdx.x >> 5;
    const int g = lane >> 4;     // half-wave group (K-offset select)
    const int r = lane & 15;     // row (A) / col (B) within tile

    const int m_base = blockIdx.y * 64 + (wave >> 1) * 16;
    const int n_base = blockIdx.x * 128 + (wave & 1) * 64;

    v8f acc[4];
#pragma unroll
    for (int s = 0; s < 4; ++s) acc[s] = (v8f)(0.0f);

    const unsigned short* aptr = A + (size_t)(m_base + r) * K + g * 8;

    for (int kk = 0; kk < K; kk += 32) {
        Frag a;
        a.q[0] = *(const uint4*)(aptr + kk);
        a.q[1] = *(const uint4*)(aptr + kk + 16);
#pragma unroll
        for (int s = 0; s < 4; ++s) {
            const unsigned short* wptr =
                W + (size_t)(n_base + s * 16 + r) * K + g * 8 + kk;
            Frag b;
            b.q[0] = *(const uint4*)(wptr);
            b.q[1] = *(const uint4*)(wptr + 16);
            acc[s] = __builtin_amdgcn_wmma_f32_16x16x32_bf16(
                false, a.v, false, b.v, (short)0, acc[s], false, false);
        }
    }

#pragma unroll
    for (int s = 0; s < 4; ++s) {
        int col = n_base + s * 16 + r;
        float bv = bias[col];
#pragma unroll
        for (int e = 0; e < 8; ++e) {
            int row = m_base + e + 8 * g;   // C/D layout: VGPR e -> M = e + 8*g
            out[(size_t)row * G3 + col] = acc[s][e] + bv;
        }
    }
}

// ---------------------------------------------------------------------------
// Device-wide barrier for the persistent recurrent kernel.
// ---------------------------------------------------------------------------
__device__ __forceinline__ void grid_barrier(unsigned int* cnt, unsigned int target) {
    __threadfence();          // make phase's global stores visible at agent scope
    __syncthreads();
    if (threadIdx.x == 0) {
        __hip_atomic_fetch_add(cnt, 1u, __ATOMIC_ACQ_REL, __HIP_MEMORY_SCOPE_AGENT);
        while (__hip_atomic_load(cnt, __ATOMIC_ACQUIRE, __HIP_MEMORY_SCOPE_AGENT) < target) {
            __builtin_amdgcn_s_sleep(2);
        }
    }
    __syncthreads();
}

// ---------------------------------------------------------------------------
// Persistent GRU recurrence, one launch per layer.
// Grid = 48 WGs (one per 64 gate-columns), 256 threads (8 waves).
// Per timestep:
//   phase 1: gh[64 x 3072] = h_bf16[64 x 1024] @ Whh^T   (WMMA bf16)
//   phase 2: WGs 0..15 compute gates + new h for their 64 hidden cols
// ---------------------------------------------------------------------------
__global__ __launch_bounds__(256) void gru_recurrent_kernel(
    const float* __restrict__ gx,            // [M x 3072], includes b_ih
    const unsigned short* __restrict__ Whh,  // [3072 x 1024] bf16
    const float* __restrict__ bhh,           // [3072]
    float* __restrict__ h,                   // [64 x 1024] fp32 state
    unsigned short* __restrict__ hbf,        // [64 x 1024] bf16 state
    float* __restrict__ gh,                  // [64 x 3072] scratch
    unsigned short* __restrict__ ybf,        // [B*T*H] bf16 output or null
    float* __restrict__ yf,                  // [B*T*H] fp32 output or null
    float* __restrict__ hidden_out,          // [64 x 1024] final hidden
    unsigned int* __restrict__ barrier_cnt,
    int T)
{
    const unsigned nwg = gridDim.x;          // 48
    const int lane = threadIdx.x & 31;
    const int wave = threadIdx.x >> 5;
    const int g = lane >> 4;
    const int r = lane & 15;
    const int wg = blockIdx.x;
    const int n0 = wg * 64;

    // wave -> (m-subtile, 2 n-subtiles) of the 64x64 WG tile
    const int m0 = (wave & 3) * 16;
    const int ncol0 = n0 + (wave >> 2) * 32;

    const unsigned short* w0base = Whh + (size_t)(ncol0 + r) * H_ + g * 8;
    const unsigned short* w1base = Whh + (size_t)(ncol0 + 16 + r) * H_ + g * 8;
    const unsigned short* abase  = hbf + (m0 + r) * H_ + g * 8;

    unsigned int phase = 0;

    for (int t = 0; t < T; ++t) {
        // ---- phase 1: recurrent GEMM tile ----
        v8f acc0 = (v8f)(0.0f), acc1 = (v8f)(0.0f);
        for (int kk = 0; kk < H_; kk += 32) {
            Frag a, b0, b1;
            a.q[0]  = *(const uint4*)(abase + kk);
            a.q[1]  = *(const uint4*)(abase + kk + 16);
            b0.q[0] = *(const uint4*)(w0base + kk);
            b0.q[1] = *(const uint4*)(w0base + kk + 16);
            b1.q[0] = *(const uint4*)(w1base + kk);
            b1.q[1] = *(const uint4*)(w1base + kk + 16);
            acc0 = __builtin_amdgcn_wmma_f32_16x16x32_bf16(
                false, a.v, false, b0.v, (short)0, acc0, false, false);
            acc1 = __builtin_amdgcn_wmma_f32_16x16x32_bf16(
                false, a.v, false, b1.v, (short)0, acc1, false, false);
        }
#pragma unroll
        for (int e = 0; e < 8; ++e) {
            int row = m0 + e + 8 * g;
            gh[row * G3 + ncol0 + r]      = acc0[e];
            gh[row * G3 + ncol0 + 16 + r] = acc1[e];
        }

        ++phase;
        grid_barrier(barrier_cnt, phase * nwg);

        // ---- phase 2: gate math on WGs 0..15 (64 hidden cols each) ----
        if (wg < 16) {
            int jl = threadIdx.x & 63;
            int j  = wg * 64 + jl;
            int rb = (threadIdx.x >> 6) * 16;
            float bh_r = bhh[j];
            float bh_z = bhh[j + H_];
            float bh_n = bhh[j + 2 * H_];
#pragma unroll 4
            for (int i = 0; i < 16; ++i) {
                int b = rb + i;
                size_t gxrow = ((size_t)b * T + t) * G3;
                float rx = gx[gxrow + j];
                float zx = gx[gxrow + H_ + j];
                float nx = gx[gxrow + 2 * H_ + j];
                float rh = gh[b * G3 + j] + bh_r;
                float zh = gh[b * G3 + H_ + j] + bh_z;
                float nh = gh[b * G3 + 2 * H_ + j] + bh_n;
                float rr = 1.0f / (1.0f + __expf(-(rx + rh)));
                float zz = 1.0f / (1.0f + __expf(-(zx + zh)));
                float nn = tanhf(nx + rr * nh);
                float hprev = h[b * H_ + j];
                float hnew  = (1.0f - zz) * nn + zz * hprev;
                h[b * H_ + j]   = hnew;
                hbf[b * H_ + j] = f32_to_bf16_rne(hnew);
                size_t yidx = ((size_t)b * T + t) * H_ + j;
                if (ybf) ybf[yidx] = f32_to_bf16_rne(hnew);
                if (yf)  yf[yidx]  = hnew;
                if (t == T - 1) hidden_out[b * H_ + j] = hnew;
            }
        }

        ++phase;
        grid_barrier(barrier_cnt, phase * nwg);
    }
}

// ---------------------------------------------------------------------------
extern "C" void kernel_launch(void* const* d_in, const int* in_sizes, int n_in,
                              void* d_out, int out_size, void* d_ws, size_t ws_size,
                              hipStream_t stream)
{
    (void)in_sizes; (void)n_in; (void)out_size; (void)ws_size;

    const float* x    = (const float*)d_in[0];
    const float* Wih0 = (const float*)d_in[1];
    const float* Whh0 = (const float*)d_in[2];
    const float* bih0 = (const float*)d_in[3];
    const float* bhh0 = (const float*)d_in[4];
    const float* Wih1 = (const float*)d_in[5];
    const float* Whh1 = (const float*)d_in[6];
    const float* bih1 = (const float*)d_in[7];
    const float* bhh1 = (const float*)d_in[8];

    char* ws = (char*)d_ws;
    size_t off = 0;
    auto alloc = [&](size_t bytes) -> void* {
        void* p = ws + off;
        off += bytes;
        off = (off + 255) & ~(size_t)255;
        return p;
    };

    float*          gx    = (float*)alloc((size_t)M_ * G3 * sizeof(float));   // reused by both layers
    unsigned short* xbf   = (unsigned short*)alloc((size_t)M_ * I_ * 2);
    unsigned short* y0bf  = (unsigned short*)alloc((size_t)M_ * H_ * 2);
    unsigned short* wih0b = (unsigned short*)alloc((size_t)G3 * I_ * 2);
    unsigned short* wih1b = (unsigned short*)alloc((size_t)G3 * H_ * 2);
    unsigned short* whh0b = (unsigned short*)alloc((size_t)G3 * H_ * 2);
    unsigned short* whh1b = (unsigned short*)alloc((size_t)G3 * H_ * 2);
    float*          hbuf  = (float*)alloc((size_t)B_ * H_ * sizeof(float));
    unsigned short* hbf   = (unsigned short*)alloc((size_t)B_ * H_ * 2);
    float*          ghbuf = (float*)alloc((size_t)B_ * G3 * sizeof(float));
    unsigned int*   cnt   = (unsigned int*)alloc(256);

    float* y1  = (float*)d_out;                       // [B,T,H]
    float* hid = (float*)d_out + (size_t)M_ * H_;     // [2,B,H]

    // --- bf16 conversions ---
    hipLaunchKernelGGL(cvt_f32_bf16_kernel, dim3(4096), dim3(256), 0, stream,
                       x, xbf, M_ * I_);
    hipLaunchKernelGGL(cvt_f32_bf16_kernel, dim3(2048), dim3(256), 0, stream,
                       Wih0, wih0b, G3 * I_);
    hipLaunchKernelGGL(cvt_f32_bf16_kernel, dim3(2048), dim3(256), 0, stream,
                       Whh0, whh0b, G3 * H_);
    hipLaunchKernelGGL(cvt_f32_bf16_kernel, dim3(2048), dim3(256), 0, stream,
                       Wih1, wih1b, G3 * H_);
    hipLaunchKernelGGL(cvt_f32_bf16_kernel, dim3(2048), dim3(256), 0, stream,
                       Whh1, whh1b, G3 * H_);

    // ================= Layer 0 =================
    // gx = x @ Wih0^T + bih0
    hipLaunchKernelGGL(gemm_bias_kernel, dim3(G3 / 128, M_ / 64), dim3(256), 0, stream,
                       xbf, wih0b, bih0, gx, I_);
    hipMemsetAsync(hbuf, 0, (size_t)B_ * H_ * sizeof(float), stream);
    hipMemsetAsync(hbf, 0, (size_t)B_ * H_ * 2, stream);
    hipMemsetAsync(cnt, 0, 256, stream);
    hipLaunchKernelGGL(gru_recurrent_kernel, dim3(48), dim3(256), 0, stream,
                       gx, whh0b, bhh0, hbuf, hbf, ghbuf,
                       y0bf, (float*)nullptr, hid, cnt, T_);

    // ================= Layer 1 =================
    // gx = y0 @ Wih1^T + bih1
    hipLaunchKernelGGL(gemm_bias_kernel, dim3(G3 / 128, M_ / 64), dim3(256), 0, stream,
                       y0bf, wih1b, bih1, gx, H_);
    hipMemsetAsync(hbuf, 0, (size_t)B_ * H_ * sizeof(float), stream);
    hipMemsetAsync(hbf, 0, (size_t)B_ * H_ * 2, stream);
    hipMemsetAsync(cnt, 0, 256, stream);
    hipLaunchKernelGGL(gru_recurrent_kernel, dim3(48), dim3(256), 0, stream,
                       gx, whh1b, bhh1, hbuf, hbf, ghbuf,
                       (unsigned short*)nullptr, y1, hid + (size_t)B_ * H_, cnt, T_);
}
